// TCATransformerEncoder_74818330296741
// MI455X (gfx1250) — compile-verified, matrix operand order checked by reading, not run on hardware
//
#include <hip/hip_runtime.h>
#include <stdint.h>

// ---------------------------------------------------------------------------
// TCA Transformer Encoder forward for MI455X (gfx1250, wave32, WMMA).
// L=2, H=8, D=512, T=1024, B=8, WIN=9.
// GEMM/attention matmuls: v_wmma_f32_16x16x32_f16 (f16 in, fp32 accumulate).
// Attention V^T tiles staged with gfx1250 async load-to-LDS.
// ---------------------------------------------------------------------------

typedef __attribute__((ext_vector_type(16))) _Float16 v16h;
typedef __attribute__((ext_vector_type(8)))  float    v8f;

union Frag { v16h h; uint32_t u[8]; };

__device__ __forceinline__ uint32_t pack2(float x, float y) {
  union { _Float16 h[2]; uint32_t u; } t;
  t.h[0] = (_Float16)x; t.h[1] = (_Float16)y;
  return t.u;
}

// CDNA5 16-bit A/B-fragment K offset for VGPR v (pairs of f16).
// lanes 0-15: K in {0..7, 16..23}; lanes 16-31: K in {8..15, 24..31}.
__device__ __forceinline__ int kkf(int v, int hi) {
  return ((v < 4) ? (2 * v) : (16 + 2 * (v - 4))) + (hi ? 8 : 0);
}

#define WMMA_F16(A, B, C) \
  __builtin_amdgcn_wmma_f32_16x16x32_f16(false, (A), false, (B), (short)0, (C), false, false)

constexpr int Dd = 512;
constexpr int Tt = 1024;
constexpr int DH = 64;   // D / H
constexpr int NB = 64;   // number of (h,b) attention blocks

// ---------------------------------------------------------------------------
// GEMM: C[M=8192, 512] = A @ W + bias (+ resid via template).  128x64 tile.
// ---------------------------------------------------------------------------
template <bool RES>
__global__ __launch_bounds__(256) void gemm512(const float* __restrict__ A,
                                               const float* __restrict__ W,
                                               const float* __restrict__ bias,
                                               const float* __restrict__ resid,
                                               float* __restrict__ C) {
  __shared__ _Float16 As[128 * 32];  // [row][k]
  __shared__ _Float16 Bs[64 * 32];   // transposed: [n][k]
  const int tid  = threadIdx.x;
  const int wave = tid >> 5, lane = tid & 31;
  const int hi = lane >> 4, ln = lane & 15;
  const int rowbase = blockIdx.x * 128;
  const int colbase = blockIdx.y * 64;

  v8f acc[4] = {};

  for (int kb = 0; kb < Dd; kb += 32) {
    // A tile: float4 loads, packed half2 stores (1024 float4 / 256 thr)
    for (int i = tid; i < 1024; i += 256) {
      const int r = i >> 3, c4 = (i & 7) * 4;
      const float4 f = *(const float4*)&A[(size_t)(rowbase + r) * Dd + kb + c4];
      uint2 pp;
      pp.x = pack2(f.x, f.y);
      pp.y = pack2(f.z, f.w);
      *(uint2*)&As[r * 32 + c4] = pp;
    }
    // W tile transposed: float4 along n, scattered f16 stores
    for (int i = tid; i < 512; i += 256) {
      const int c = i >> 4, n4 = (i & 15) * 4;
      const float4 f = *(const float4*)&W[(size_t)(kb + c) * Dd + colbase + n4];
      Bs[(n4 + 0) * 32 + c] = (_Float16)f.x;
      Bs[(n4 + 1) * 32 + c] = (_Float16)f.y;
      Bs[(n4 + 2) * 32 + c] = (_Float16)f.z;
      Bs[(n4 + 3) * 32 + c] = (_Float16)f.w;
    }
    __syncthreads();

    Frag a;
    const int arow = wave * 16 + ln;
#pragma unroll
    for (int v = 0; v < 8; ++v)
      a.u[v] = *(const uint32_t*)&As[arow * 32 + kkf(v, hi)];

#pragma unroll
    for (int t = 0; t < 4; ++t) {
      Frag b;
      const int ncol = t * 16 + ln;
#pragma unroll
      for (int v = 0; v < 8; ++v)
        b.u[v] = *(const uint32_t*)&Bs[ncol * 32 + kkf(v, hi)];
      acc[t] = WMMA_F16(a.h, b.h, acc[t]);
    }
    __syncthreads();
  }

#pragma unroll
  for (int t = 0; t < 4; ++t) {
#pragma unroll
    for (int v = 0; v < 8; ++v) {
      const int row = rowbase + wave * 16 + v + 8 * hi;
      const int col = colbase + t * 16 + ln;
      const size_t o = (size_t)row * Dd + col;
      float val = acc[t][v] + bias[col];
      if (RES) val += resid[o];
      C[o] = val;
    }
  }
}

// ---------------------------------------------------------------------------
// prep: f16 copies of Q,K (natural layout) and V pre-transposed per block:
// vt16[blk][dh][key]  (blk = flat/65536, key = (flat>>6)&1023, dh = flat&63)
// ---------------------------------------------------------------------------
__global__ __launch_bounds__(256) void prep_attn(const float* __restrict__ q,
                                                 const float* __restrict__ k,
                                                 const float* __restrict__ v,
                                                 _Float16* __restrict__ q16,
                                                 _Float16* __restrict__ k16,
                                                 _Float16* __restrict__ vt16) {
  const size_t i = (size_t)blockIdx.x * 256 + threadIdx.x;
  q16[i] = (_Float16)q[i];
  k16[i] = (_Float16)k[i];
  const int dh  = (int)(i & 63);
  const int key = (int)((i >> 6) & 1023);
  const size_t blk = i >> 16;
  vt16[(blk << 16) + (size_t)dh * 1024 + key] = (_Float16)v[i];
}

// ---------------------------------------------------------------------------
// Fused dual-branch attention.  Grid (64 blocks, 32 q-tiles), 64 thr (2 waves).
// ---------------------------------------------------------------------------
__global__ __launch_bounds__(64) void attn_kernel(const _Float16* __restrict__ Q16,
                                                  const _Float16* __restrict__ K16,
                                                  const _Float16* __restrict__ Vt16,
                                                  const float* __restrict__ Qf,
                                                  const float* __restrict__ Kf,
                                                  const float* __restrict__ Vf,
                                                  const float* __restrict__ alpha,
                                                  const float* __restrict__ adjw,
                                                  const float* __restrict__ adjb,
                                                  float* __restrict__ Out,
                                                  int layer) {
  __shared__ _Float16 Vt[64 * 64];      // [dh][key] for current 64-key chunk
  __shared__ _Float16 Pl[2][16 * 64];   // per-wave P tile [row][key]
  __shared__ float    Gl[32 * 64];      // global-branch output
  __shared__ float    lw[32][12];       // local band weights
  __shared__ int      lof[32], lcnt[32];

  const int blk = blockIdx.x;
  const int qb  = blockIdx.y * 32;
  const int tid = threadIdx.x;
  const int wave = tid >> 5, lane = tid & 31;
  const int hi = lane >> 4, ln = lane & 15;
  const size_t base = (size_t)blk * Tt * DH;
  const float scale = 0.044194173824159216f;  // 1/sqrt(512)
  const float aw = adjw[0], ab = adjb[0];

  // Q fragments (dh chunks 0..31 and 32..63) -- direct dword loads from f16
  Frag qf[2];
  {
    const int qrow = qb + wave * 16 + ln;
#pragma unroll
    for (int c = 0; c < 2; ++c)
#pragma unroll
      for (int v = 0; v < 8; ++v)
        qf[c].u[v] =
            *(const uint32_t*)&Q16[base + (size_t)qrow * DH + c * 32 + kkf(v, hi)];
  }

  v8f acc[4] = {};
  float m[8], l[8];
#pragma unroll
  for (int v = 0; v < 8; ++v) { m[v] = -3.0e38f; l[v] = 0.f; }

  for (int kc = 0; kc < Tt; kc += 64) {
    // stage V^T chunk via gfx1250 async global->LDS copies (no VGPR round trip)
    {
      const _Float16* gsrc = Vt16 + base + (size_t)tid * 1024 + kc;  // row dh=tid
      const uint32_t lds0 = (uint32_t)(uintptr_t)(&Vt[tid * 64]);
#pragma unroll
      for (int j = 0; j < 8; ++j) {  // 64 f16 = 128 B = 8 x b128
        asm volatile("global_load_async_to_lds_b128 %0, %1, off"
                     :: "v"(lds0 + j * 16), "v"(gsrc + j * 8)
                     : "memory");
      }
      asm volatile("s_wait_asynccnt 0x0" ::: "memory");
    }
    __syncthreads();

    // S = scale * Q K^T + adj
    float sf[4][8];
#pragma unroll
    for (int t = 0; t < 4; ++t) {
      Frag kf0, kf1;
      const int key = kc + t * 16 + ln;
#pragma unroll
      for (int v = 0; v < 8; ++v) {
        const int k0 = kkf(v, hi);
        kf0.u[v] = *(const uint32_t*)&K16[base + (size_t)key * DH + k0];
        kf1.u[v] = *(const uint32_t*)&K16[base + (size_t)key * DH + 32 + k0];
      }
      v8f s = {};
      s = WMMA_F16(qf[0].h, kf0.h, s);
      s = WMMA_F16(qf[1].h, kf1.h, s);
#pragma unroll
      for (int v = 0; v < 8; ++v) {
        const int row = qb + wave * 16 + v + 8 * hi;
        const int col = kc + t * 16 + ln;
        const float d = fabsf((float)(row - col));
        const float adj = __expf(-fabsf(aw * d * d - ab));
        sf[t][v] = s[v] * scale + adj;
      }
    }

    // online softmax (global branch); rows r and r+8 live in disjoint halves
#pragma unroll
    for (int v = 0; v < 8; ++v) {
      float rm = fmaxf(fmaxf(sf[0][v], sf[1][v]), fmaxf(sf[2][v], sf[3][v]));
#pragma unroll
      for (int xm = 1; xm < 16; xm <<= 1) rm = fmaxf(rm, __shfl_xor(rm, xm, 32));
      const float mn = fmaxf(m[v], rm);
      const float corr = __expf(m[v] - mn);
      float rs = 0.f;
#pragma unroll
      for (int t = 0; t < 4; ++t) {
        const float p = __expf(sf[t][v] - mn);
        sf[t][v] = p;
        rs += p;
      }
#pragma unroll
      for (int xm = 1; xm < 16; xm <<= 1) rs += __shfl_xor(rs, xm, 32);
      l[v] = l[v] * corr + rs;
      m[v] = mn;
#pragma unroll
      for (int t = 0; t < 4; ++t) acc[t][v] *= corr;
    }

    // P -> LDS (row-major, contraction pairs contiguous)
#pragma unroll
    for (int t = 0; t < 4; ++t)
#pragma unroll
      for (int v = 0; v < 8; ++v)
        Pl[wave][(v + 8 * hi) * 64 + t * 16 + ln] = (_Float16)sf[t][v];
    __syncthreads();

    // acc += P @ Vchunk
#pragma unroll
    for (int c2 = 0; c2 < 2; ++c2) {
      Frag pf;
#pragma unroll
      for (int v = 0; v < 8; ++v)
        pf.u[v] = *(const uint32_t*)&Pl[wave][ln * 64 + c2 * 32 + kkf(v, hi)];
#pragma unroll
      for (int t = 0; t < 4; ++t) {
        Frag vf;
#pragma unroll
        for (int v = 0; v < 8; ++v)
          vf.u[v] = *(const uint32_t*)&Vt[(t * 16 + ln) * 64 + c2 * 32 + kkf(v, hi)];
        acc[t] = WMMA_F16(pf.h, vf.h, acc[t]);
      }
    }
    __syncthreads();
  }

  // global branch result -> LDS
#pragma unroll
  for (int t = 0; t < 4; ++t)
#pragma unroll
    for (int v = 0; v < 8; ++v)
      Gl[(wave * 16 + v + 8 * hi) * 64 + t * 16 + ln] = acc[t][v] / l[v];
  __syncthreads();

  // local branch: band [q-4, q+4] clipped to [0, T-1] (fp32 inputs)
  const float* Qb = Qf + base;
  const float* Kb = Kf + base;
  const float* Vb = Vf + base;
  if (tid < 32) {
    const int row = tid;
    const int qg = qb + row;
    const int lo = (qg - 4 < 0) ? 0 : qg - 4;
    const int hi2 = (qg + 4 > Tt - 1) ? Tt - 1 : qg + 4;
    const int cnt = hi2 - lo + 1;
    float lg[9];
    float mx = -3.0e38f;
    for (int j = 0; j < cnt; ++j) {
      const int col = lo + j;
      float dot = 0.f;
      for (int dd = 0; dd < DH; ++dd)
        dot += Qb[(size_t)qg * DH + dd] * Kb[(size_t)col * DH + dd];
      const float d = fabsf((float)(qg - col));
      lg[j] = dot * scale + __expf(-fabsf(aw * d * d - ab));
      mx = fmaxf(mx, lg[j]);
    }
    float sm = 0.f;
    for (int j = 0; j < cnt; ++j) { lg[j] = __expf(lg[j] - mx); sm += lg[j]; }
    const float inv = 1.f / sm;
    for (int j = 0; j < cnt; ++j) lw[row][j] = lg[j] * inv;
    lof[row] = lo;
    lcnt[row] = cnt;
  }
  __syncthreads();

  const float a_sig = 1.f / (1.f + __expf(-alpha[layer]));
  for (int i = tid; i < 32 * 64; i += 64) {
    const int row = i >> 6, dh = i & 63;
    const int lo = lof[row], cnt = lcnt[row];
    float lcl = 0.f;
    for (int j = 0; j < cnt; ++j)
      lcl += lw[row][j] * Vb[(size_t)(lo + j) * DH + dh];
    Out[base + (size_t)(qb + row) * DH + dh] = a_sig * Gl[i] + (1.f - a_sig) * lcl;
  }
}

// ---------------------------------------------------------------------------
// power-norm + F.normalize(dim=1): per (b,d) column norm over T.
// ---------------------------------------------------------------------------
__global__ __launch_bounds__(256) void powernorm_kernel(const float* __restrict__ X,
                                                        float* __restrict__ Y) {
  __shared__ float red[4][64];
  __shared__ float inv[64];
  const int b = blockIdx.x >> 3;
  const int dbase = (blockIdx.x & 7) * 64;
  const int tid = threadIdx.x;
  const int dl = tid & 63, rl = tid >> 6;
  const int d = dbase + dl;
  float ss = 0.f;
  for (int t = rl; t < Tt; t += 4) {
    const size_t idx = ((size_t)b * Tt + t) * Dd + d;
    const float v = X[idx];
    const float p = sqrtf(fmaxf(v, 0.f)) - sqrtf(fmaxf(-v, 0.f));
    Y[idx] = p;
    ss += p * p;
  }
  red[rl][dl] = ss;
  __syncthreads();
  if (tid < 64) {
    const float s = red[0][tid] + red[1][tid] + red[2][tid] + red[3][tid];
    inv[tid] = 1.f / fmaxf(sqrtf(s), 1e-12f);
  }
  __syncthreads();
  const float iv = inv[dl];
  for (int t = rl; t < Tt; t += 4)
    Y[((size_t)b * Tt + t) * Dd + d] *= iv;
}

// ---------------------------------------------------------------------------
// Row LayerNorm over D=512.  grid 8192, 128 threads.
// ---------------------------------------------------------------------------
__global__ __launch_bounds__(128) void layernorm_kernel(const float* __restrict__ X,
                                                        const float* __restrict__ g,
                                                        const float* __restrict__ bta,
                                                        float* __restrict__ Y) {
  __shared__ float red[128];
  const int row = blockIdx.x;
  const int tid = threadIdx.x;
  const float* xr = X + (size_t)row * Dd;
  float v[4], s = 0.f;
#pragma unroll
  for (int i = 0; i < 4; ++i) { v[i] = xr[tid + i * 128]; s += v[i]; }
  red[tid] = s;
  __syncthreads();
  for (int off = 64; off; off >>= 1) {
    if (tid < off) red[tid] += red[tid + off];
    __syncthreads();
  }
  const float mu = red[0] * (1.f / 512.f);
  __syncthreads();
  float s2 = 0.f;
#pragma unroll
  for (int i = 0; i < 4; ++i) { const float dv = v[i] - mu; s2 += dv * dv; }
  red[tid] = s2;
  __syncthreads();
  for (int off = 64; off; off >>= 1) {
    if (tid < off) red[tid] += red[tid + off];
    __syncthreads();
  }
  const float rstd = rsqrtf(red[0] * (1.f / 512.f) + 1e-5f);
#pragma unroll
  for (int i = 0; i < 4; ++i) {
    const int c = tid + i * 128;
    Y[(size_t)row * Dd + c] = (v[i] - mu) * rstd * g[c] + bta[c];
  }
}

// ---------------------------------------------------------------------------
// [T,B,D] <-> [B,T,D] transposes (flat i = t*4096 + b*512 + d).
// ---------------------------------------------------------------------------
__global__ void t_in(const float* __restrict__ X, float* __restrict__ Y) {
  const size_t i = (size_t)blockIdx.x * 256 + threadIdx.x;
  const int d = (int)(i & 511);
  const int b = (int)((i >> 9) & 7);
  const int t = (int)(i >> 12);
  Y[((size_t)b * Tt + t) * Dd + d] = X[i];
}

__global__ void t_out(const float* __restrict__ X, float* __restrict__ Y) {
  const size_t i = (size_t)blockIdx.x * 256 + threadIdx.x;
  const int d = (int)(i & 511);
  const int b = (int)((i >> 9) & 7);
  const int t = (int)(i >> 12);
  Y[i] = X[((size_t)b * Tt + t) * Dd + d];
}

// ---------------------------------------------------------------------------
extern "C" void kernel_launch(void* const* d_in, const int* in_sizes, int n_in,
                              void* d_out, int out_size, void* d_ws, size_t ws_size,
                              hipStream_t stream) {
  (void)in_sizes; (void)n_in; (void)out_size; (void)ws_size;
  const float* x     = (const float*)d_in[0];
  const float* Wq    = (const float*)d_in[1];
  const float* bq    = (const float*)d_in[2];
  const float* Wk    = (const float*)d_in[3];
  const float* bk    = (const float*)d_in[4];
  const float* Wv    = (const float*)d_in[5];
  const float* bv    = (const float*)d_in[6];
  const float* Wo    = (const float*)d_in[7];
  const float* bo    = (const float*)d_in[8];
  const float* alpha = (const float*)d_in[9];
  const float* lng   = (const float*)d_in[10];
  const float* lnb   = (const float*)d_in[11];
  const float* adjw  = (const float*)d_in[12];
  const float* adjb  = (const float*)d_in[13];
  float* out = (float*)d_out;

  const size_t S = (size_t)8 * 1024 * 512;  // one [B,T,D] buffer (elements)
  float* ws   = (float*)d_ws;
  float* xb   = ws;
  float* q    = ws + S;
  float* k    = ws + 2 * S;
  float* v    = ws + 3 * S;
  float* attn = ws + 4 * S;
  float* pn   = q;  // q dead after attention
  float* y    = k;  // k dead after attention
  _Float16* h16  = (_Float16*)(ws + 5 * S);
  _Float16* q16  = h16;
  _Float16* k16  = h16 + S;
  _Float16* vt16 = h16 + 2 * S;

  t_in<<<16384, 256, 0, stream>>>(x, xb);

  for (int L = 0; L < 2; ++L) {
    const int WOFF = L * Dd * Dd;
    const int BOFF = L * Dd;
    gemm512<false><<<dim3(64, 8), 256, 0, stream>>>(xb, Wq + WOFF, bq + BOFF, nullptr, q);
    gemm512<false><<<dim3(64, 8), 256, 0, stream>>>(xb, Wk + WOFF, bk + BOFF, nullptr, k);
    gemm512<false><<<dim3(64, 8), 256, 0, stream>>>(xb, Wv + WOFF, bv + BOFF, nullptr, v);
    prep_attn<<<16384, 256, 0, stream>>>(q, k, v, q16, k16, vt16);
    attn_kernel<<<dim3(NB, 32), 64, 0, stream>>>(q16, k16, vt16, q, k, v,
                                                 alpha, adjw, adjb, attn, L);
    powernorm_kernel<<<64, 256, 0, stream>>>(attn, pn);
    gemm512<true><<<dim3(64, 8), 256, 0, stream>>>(pn, Wo + WOFF, bo + BOFF, xb, y);
    layernorm_kernel<<<8192, 128, 0, stream>>>(y, lng + BOFF, lnb + BOFF, xb);
  }

  t_out<<<16384, 256, 0, stream>>>(xb, out);
}
